// RNN_70351564309228
// MI455X (gfx1250) — matrix operand order repaired and checked by reading, not compiled, not run on hardware
//
#include <hip/hip_runtime.h>
#include <math.h>

// ---------------------------------------------------------------------------
// LSTM unroll: T=128, B=32, D=1024, H=1024.
// Strategy: bf16 WMMA (v_wmma_f32_16x16x32_bf16) with fp32 accum; weights
// converted+transposed once to bf16 [4H,K] so B-operand lanes read contiguous
// K; recurrence handled by 128 stream-ordered step launches (implicit global
// sync); all weights stay L2-resident (16 MB bf16 << 192 MB L2).
// This revision: dual accumulators per wave to break the WMMA RAW chain
// (2x matrix-pipe ILP in the latency-bound sequential region).
// ---------------------------------------------------------------------------

typedef __attribute__((ext_vector_type(16))) __bf16 bf16x16;
typedef __attribute__((ext_vector_type(8)))  __bf16 bf16x8;
typedef __attribute__((ext_vector_type(8)))  float  f32x8;

#define T_  128
#define B_  32
#define D_  1024
#define H_  1024
#define G4H (4 * H_)

// --- one-time conversion kernels -------------------------------------------

__global__ void to_bf16_kernel(const float* __restrict__ src,
                               __bf16* __restrict__ dst, int n) {
  int stride = gridDim.x * blockDim.x;
  for (int i = blockIdx.x * blockDim.x + threadIdx.x; i < n; i += stride)
    dst[i] = (__bf16)src[i];
}

// src [K,N] fp32 row-major -> dst [N,K] bf16 row-major (transpose+convert)
__global__ void transpose_to_bf16_kernel(const float* __restrict__ src,
                                         __bf16* __restrict__ dst,
                                         int K, int N) {
  int stride = gridDim.x * blockDim.x;
  int total = K * N;
  for (int i = blockIdx.x * blockDim.x + threadIdx.x; i < total; i += stride) {
    int n = i / K;
    int k = i - n * K;
    dst[i] = (__bf16)src[(size_t)k * N + n];
  }
}

__global__ void init_state_kernel(__bf16* __restrict__ h,
                                  float* __restrict__ c, int n) {
  int stride = gridDim.x * blockDim.x;
  for (int i = blockIdx.x * blockDim.x + threadIdx.x; i < n; i += stride) {
    h[i] = (__bf16)0.0f;
    c[i] = 0.0f;
  }
}

// --- WMMA operand loaders ---------------------------------------------------
// A (16x32 bf16, M=rows): lane<16 holds K {0..7}U{16..23}, lane>=16 holds
// K {8..15}U{24..31}; caller pre-applies the +0/+8 half offset.
__device__ __forceinline__ bf16x16 load_a_frag(const __bf16* p) {
  bf16x8 lo = *reinterpret_cast<const bf16x8*>(p);
  bf16x8 hi = *reinterpret_cast<const bf16x8*>(p + 16);
  bf16x16 r;
#pragma unroll
  for (int i = 0; i < 8; ++i) { r[i] = lo[i]; r[i + 8] = hi[i]; }
  return r;
}

// B (32x16 bf16, N=cols): lane<16 holds K {0..7}U{16..23}, lane>=16 holds
// K {8..15}U{24..31} of column n; caller pre-applies +0/+8.
__device__ __forceinline__ bf16x16 load_b_frag(const __bf16* p) {
  bf16x8 lo = *reinterpret_cast<const bf16x8*>(p);
  bf16x8 hi = *reinterpret_cast<const bf16x8*>(p + 16);
  bf16x16 r;
#pragma unroll
  for (int i = 0; i < 8; ++i) { r[i] = lo[i]; r[i + 8] = hi[i]; }
  return r;
}

// Dual-accumulator K reduction over one (A row-tile, B col-tile) pair.
// Two independent WMMA dependency chains: chunks alternate acc0/acc1.
__device__ __forceinline__ void gemm_k_loop(const __bf16* __restrict__ arow,
                                            const __bf16* __restrict__ brow,
                                            int Klen, f32x8& acc0, f32x8& acc1) {
#pragma unroll 2
  for (int k0 = 0; k0 < Klen; k0 += 64) {
    __builtin_prefetch(brow + k0 + 512, 0, 1);
    bf16x16 a0 = load_a_frag(arow + k0);
    bf16x16 b0 = load_b_frag(brow + k0);
    bf16x16 a1 = load_a_frag(arow + k0 + 32);
    bf16x16 b1 = load_b_frag(brow + k0 + 32);
    acc0 = __builtin_amdgcn_wmma_f32_16x16x32_bf16(
        false, a0, false, b0, (short)0, acc0, false, false);
    acc1 = __builtin_amdgcn_wmma_f32_16x16x32_bf16(
        false, a1, false, b1, (short)0, acc1, false, false);
  }
}

// --- per-timestep kernel ----------------------------------------------------
// grid = 64 blocks (16 hidden units each), block = 256 threads = 8 waves.
// Wave w: gate g = w&3, batch half m0 = (w>>2)*16, two 16x16 f32 accumulators
// (independent K streams), K loop over D (x@Wx) then H (h@Wh).
__global__ __launch_bounds__(256) void lstm_step_kernel(
    const __bf16* __restrict__ xbf,   // [T*B, D]
    const __bf16* __restrict__ WxT,   // [4H, D]
    const __bf16* __restrict__ WhT,   // [4H, H]
    const float*  __restrict__ bias,  // [4H]
    __bf16* __restrict__ hbf,         // [B, H] recurrent state (bf16)
    float*  __restrict__ cst,         // [B, H] cell state
    float*  __restrict__ hs,          // [T, B, H] output
    int t) {
  __shared__ float gates[B_][64 + 1];

  const int tid  = threadIdx.x;
  const int lane = tid & 31;
  const int w    = tid >> 5;
  const int n0   = blockIdx.x * 16;     // first hidden unit of this block
  const int g    = w & 3;               // gate index 0..3 (i,f,g,o order)
  const int m0   = (w >> 2) * 16;       // batch-row tile base (0 or 16)
  const int lrow = lane & 15;
  const int ah   = (lane < 16) ? 0 : 8;   // A-fragment K half offset
  const int bh   = (lane < 16) ? 0 : 8;   // B-fragment K half offset

  const int gcol = g * H_ + n0 + lrow;  // weight row in transposed layout

  f32x8 acc0 = {};
  f32x8 acc1 = {};

  // ---- x_t @ Wx contribution (K = D) ----
  gemm_k_loop(xbf + (size_t)(t * B_ + m0 + lrow) * D_ + ah,
              WxT + (size_t)gcol * D_ + bh, D_, acc0, acc1);

  // ---- h_{t-1} @ Wh contribution (K = H) ----
  gemm_k_loop(hbf + (size_t)(m0 + lrow) * H_ + ah,
              WhT + (size_t)gcol * H_ + bh, H_, acc0, acc1);

  // ---- stage accumulator to LDS (C/D layout: elem e -> M = e + hi*8) ----
  const int mhi = (lane < 16) ? 0 : 8;
#pragma unroll
  for (int e = 0; e < 8; ++e)
    gates[m0 + mhi + e][g * 16 + lrow] = acc0[e] + acc1[e];

  __syncthreads();

  // ---- LSTM cell: 32 batches x 16 units = 512 elements ----
  for (int idx = tid; idx < B_ * 16; idx += 256) {
    int bi = idx >> 4;
    int u  = idx & 15;
    int col = n0 + u;
    float gi = gates[bi][0 * 16 + u] + bias[0 * H_ + col];
    float gf = gates[bi][1 * 16 + u] + bias[1 * H_ + col];
    float gg = gates[bi][2 * 16 + u] + bias[2 * H_ + col];
    float go = gates[bi][3 * 16 + u] + bias[3 * H_ + col];
    float si = 1.0f / (1.0f + __expf(-gi));
    float sf = 1.0f / (1.0f + __expf(-gf));
    float so = 1.0f / (1.0f + __expf(-go));
    float tg = tanhf(gg);
    float cold = cst[bi * H_ + col];
    float cnew = sf * cold + si * tg;
    float hnew = so * tanhf(cnew);
    cst[bi * H_ + col] = cnew;
    hbf[bi * H_ + col] = (__bf16)hnew;
    hs[(size_t)t * B_ * H_ + (size_t)bi * H_ + col] = hnew;
  }
}

// --- launch ----------------------------------------------------------------

extern "C" void kernel_launch(void* const* d_in, const int* in_sizes, int n_in,
                              void* d_out, int out_size, void* d_ws,
                              size_t ws_size, hipStream_t stream) {
  const float* x  = (const float*)d_in[0];  // [T, B, D]
  const float* Wx = (const float*)d_in[1];  // [D, 4H]
  const float* Wh = (const float*)d_in[2];  // [H, 4H]
  const float* b  = (const float*)d_in[3];  // [4H]
  float* hs = (float*)d_out;                // [T, B, H]

  char* ws = (char*)d_ws;
  // workspace layout (bytes):
  //   [0, 8M)     x as bf16             [T*B, D]
  //   [8M, 16M)   WxT bf16 (transposed) [4H, D]
  //   [16M, 24M)  WhT bf16 (transposed) [4H, H]
  //   [24M, ..)   h state bf16 [B,H] (64 KB), then c state fp32 [B,H] (128 KB)
  __bf16* xbf = (__bf16*)(ws);
  __bf16* WxT = (__bf16*)(ws + (size_t)8 * 1024 * 1024);
  __bf16* WhT = (__bf16*)(ws + (size_t)16 * 1024 * 1024);
  __bf16* hbf = (__bf16*)(ws + (size_t)24 * 1024 * 1024);
  float*  cst = (float*)(ws + (size_t)24 * 1024 * 1024 + 64 * 1024);

  to_bf16_kernel<<<2048, 256, 0, stream>>>(x, xbf, T_ * B_ * D_);
  transpose_to_bf16_kernel<<<2048, 256, 0, stream>>>(Wx, WxT, D_, G4H);
  transpose_to_bf16_kernel<<<2048, 256, 0, stream>>>(Wh, WhT, H_, G4H);
  init_state_kernel<<<128, 256, 0, stream>>>(hbf, cst, B_ * H_);

  for (int t = 0; t < T_; ++t) {
    lstm_step_kernel<<<64, 256, 0, stream>>>(xbf, WxT, WhT, b, hbf, cst, hs, t);
  }
}
